// BinaryLinearUnit_16587163697152
// MI455X (gfx1250) — 8-bit matrix operand order repaired and checked by reading, compiled, not run on hardware
//
#include <hip/hip_runtime.h>

typedef __attribute__((ext_vector_type(16))) int   v16i;
typedef __attribute__((ext_vector_type(8)))  float v8f;

constexpr int BDIM = 8192;
constexpr int KDIM = 4096;
constexpr int NDIM = 4096;

constexpr int TM = 128;
constexpr int TN = 128;
constexpr int TK = 128;
constexpr int LDS_STRIDE = TK + 16;

__global__ __launch_bounds__(256) void binarize_fp8_kernel(
    const float* __restrict__ src, unsigned int* __restrict__ dst, int n4) {
  int stride = gridDim.x * blockDim.x;
  for (int i = blockIdx.x * blockDim.x + threadIdx.x; i < n4; i += stride) {
    float4 v = reinterpret_cast<const float4*>(src)[i];
    unsigned b0 = v.x > 0.f ? 0x38u : (v.x < 0.f ? 0xB8u : 0x00u);
    unsigned b1 = v.y > 0.f ? 0x38u : (v.y < 0.f ? 0xB8u : 0x00u);
    unsigned b2 = v.z > 0.f ? 0x38u : (v.z < 0.f ? 0xB8u : 0x00u);
    unsigned b3 = v.w > 0.f ? 0x38u : (v.w < 0.f ? 0xB8u : 0x00u);
    dst[i] = b0 | (b1 << 8) | (b2 << 16) | (b3 << 24);
  }
}

__global__ __launch_bounds__(256) void bgemm_fp8_wmma_kernel(
    const unsigned char* __restrict__ A,
    const unsigned char* __restrict__ W,
    float* __restrict__ Y)
{
  constexpr int ASZ = TM * LDS_STRIDE;
  constexpr int WSZ = TN * LDS_STRIDE;
  __shared__ __align__(16) unsigned char a_sh[2 * ASZ];
  __shared__ __align__(16) unsigned char w_sh[2 * WSZ];

  const int tid  = threadIdx.x;
  const int lane = tid & 31;
  const int wave = tid >> 5;
  const int wm   = wave & 3;
  const int wn   = wave >> 2;
  const int mh   = lane >> 4;
  const int ml   = lane & 15;

  const size_t m0 = (size_t)blockIdx.y * TM;
  const size_t n0 = (size_t)blockIdx.x * TN;

  const unsigned char* a_gp[4];
  const unsigned char* w_gp[4];
  unsigned a_l0[4], w_l0[4];
#pragma unroll
  for (int i = 0; i < 4; ++i) {
    int c   = tid + i * 256;
    int r   = c >> 3;
    int col = (c & 7) << 4;
    a_gp[i] = A + (m0 + (size_t)r) * KDIM + col;
    w_gp[i] = W + (n0 + (size_t)r) * KDIM + col;
    a_l0[i] = (unsigned)(uintptr_t)a_sh + (unsigned)(r * LDS_STRIDE + col);
    w_l0[i] = (unsigned)(uintptr_t)w_sh + (unsigned)(r * LDS_STRIDE + col);
  }

  v8f acc[2][4];
  v8f vzero = {};
#pragma unroll
  for (int tm = 0; tm < 2; ++tm)
#pragma unroll
    for (int tn = 0; tn < 4; ++tn) acc[tm][tn] = vzero;

  constexpr int NT = KDIM / TK;

#pragma unroll
  for (int i = 0; i < 4; ++i) {
    asm volatile("global_load_async_to_lds_b128 %0, %1, off"
                 :: "v"(a_l0[i]), "v"(a_gp[i]) : "memory");
    asm volatile("global_load_async_to_lds_b128 %0, %1, off"
                 :: "v"(w_l0[i]), "v"(w_gp[i]) : "memory");
  }
#pragma unroll
  for (int i = 0; i < 4; ++i) {
    asm volatile("global_load_async_to_lds_b128 %0, %1, off"
                 :: "v"(a_l0[i] + ASZ), "v"(a_gp[i] + TK) : "memory");
    asm volatile("global_load_async_to_lds_b128 %0, %1, off"
                 :: "v"(w_l0[i] + WSZ), "v"(w_gp[i] + TK) : "memory");
  }
#pragma unroll
  for (int i = 0; i < 4; ++i) { a_gp[i] += 2 * TK; w_gp[i] += 2 * TK; }

  for (int k = 0; k < NT; ++k) {
    const int cur = k & 1;

    if (k + 1 < NT) asm volatile("s_wait_asynccnt 0x8" ::: "memory");
    else            asm volatile("s_wait_asynccnt 0x0" ::: "memory");
    __syncthreads();

    const unsigned char* ash = a_sh + cur * ASZ;
    const unsigned char* wsh = w_sh + cur * WSZ;

    v16i afrag[2];
#pragma unroll
    for (int tm = 0; tm < 2; ++tm) {
      const unsigned char* base = &ash[(wm * 32 + tm * 16 + ml) * LDS_STRIDE + mh * 16];
      v16i a;
#pragma unroll
      for (int q = 0; q < 4; ++q) {
        int4 t = *reinterpret_cast<const int4*>(base + q * 32);
        a[4 * q + 0] = t.x;
        a[4 * q + 1] = t.y;
        a[4 * q + 2] = t.z;
        a[4 * q + 3] = t.w;
      }
      afrag[tm] = a;
    }

    v16i bfrag[4];
#pragma unroll
    for (int tn = 0; tn < 4; ++tn) {
      const unsigned char* base = &wsh[(wn * 64 + tn * 16 + ml) * LDS_STRIDE + mh * 16];
      v16i b;
#pragma unroll
      for (int q = 0; q < 4; ++q) {
        int4 t = *reinterpret_cast<const int4*>(base + q * 32);
        b[4 * q + 0] = t.x;
        b[4 * q + 1] = t.y;
        b[4 * q + 2] = t.z;
        b[4 * q + 3] = t.w;
      }
      bfrag[tn] = b;
    }

#pragma unroll
    for (int tm = 0; tm < 2; ++tm)
#pragma unroll
      for (int tn = 0; tn < 4; ++tn)
        acc[tm][tn] = __builtin_amdgcn_wmma_f32_16x16x128_fp8_fp8(
            afrag[tm], bfrag[tn], (short)0, acc[tm][tn], false, false);

    __syncthreads();

    if (k + 2 < NT) {
      const unsigned aoff = (unsigned)(cur * ASZ);
      const unsigned woff = (unsigned)(cur * WSZ);
#pragma unroll
      for (int i = 0; i < 4; ++i) {
        asm volatile("global_load_async_to_lds_b128 %0, %1, off"
                     :: "v"(a_l0[i] + aoff), "v"(a_gp[i]) : "memory");
        asm volatile("global_load_async_to_lds_b128 %0, %1, off"
                     :: "v"(w_l0[i] + woff), "v"(w_gp[i]) : "memory");
      }
#pragma unroll
      for (int i = 0; i < 4; ++i) { a_gp[i] += TK; w_gp[i] += TK; }
    }
  }

#pragma unroll
  for (int tm = 0; tm < 2; ++tm)
#pragma unroll
    for (int tn = 0; tn < 4; ++tn)
#pragma unroll
      for (int v = 0; v < 8; ++v) {
        size_t m = m0 + (size_t)(wm * 32 + tm * 16 + v + 8 * mh);
        size_t n = n0 + (size_t)(wn * 64 + tn * 16 + ml);
        Y[m * NDIM + n] = acc[tm][tn][v];
      }
}

__global__ __launch_bounds__(256) void stats_kernel(
    const float* __restrict__ Y, float* __restrict__ mean, float* __restrict__ rstd) {
  __shared__ double s_sum[256];
  __shared__ double s_sq[256];

  const int c     = blockIdx.x * 32 + (threadIdx.x & 31);
  const int slice = threadIdx.x >> 5;

  double sum = 0.0, sq = 0.0;
  for (int r = slice; r < BDIM; r += 8) {
    double v = (double)Y[(size_t)r * NDIM + c];
    sum += v;
    sq  += v * v;
  }
  s_sum[threadIdx.x] = sum;
  s_sq[threadIdx.x]  = sq;
  __syncthreads();

  if (threadIdx.x < 32) {
#pragma unroll
    for (int s = 1; s < 8; ++s) {
      sum += s_sum[threadIdx.x + 32 * s];
      sq  += s_sq[threadIdx.x + 32 * s];
    }
    double m   = sum / (double)BDIM;
    double var = sq / (double)BDIM - m * m;
    mean[c] = (float)m;
    rstd[c] = rsqrtf((float)var + 1e-5f);
  }
}

__global__ __launch_bounds__(256) void norm_kernel(
    float* __restrict__ Y, const float* __restrict__ mean, const float* __restrict__ rstd,
    const float* __restrict__ gamma, const float* __restrict__ beta) {
  size_t i = (size_t)blockIdx.x * 256 + threadIdx.x;
  float4 v = reinterpret_cast<float4*>(Y)[i];
  int c = (int)((i * 4) & (size_t)(NDIM - 1));
  v.x = (v.x - mean[c + 0]) * rstd[c + 0] * gamma[c + 0] + beta[c + 0];
  v.y = (v.y - mean[c + 1]) * rstd[c + 1] * gamma[c + 1] + beta[c + 1];
  v.z = (v.z - mean[c + 2]) * rstd[c + 2] * gamma[c + 2] + beta[c + 2];
  v.w = (v.w - mean[c + 3]) * rstd[c + 3] * gamma[c + 3] + beta[c + 3];
  reinterpret_cast<float4*>(Y)[i] = v;
}

extern "C" void kernel_launch(void* const* d_in, const int* in_sizes, int n_in,
                              void* d_out, int out_size, void* d_ws, size_t ws_size,
                              hipStream_t stream) {
  const float* x     = (const float*)d_in[0];
  const float* w     = (const float*)d_in[1];
  const float* gamma = (const float*)d_in[2];
  const float* beta  = (const float*)d_in[3];
  float* y = (float*)d_out;

  unsigned char* xs  = (unsigned char*)d_ws;
  unsigned char* wss = xs + (size_t)BDIM * KDIM;
  float* mean = (float*)(wss + (size_t)NDIM * KDIM);
  float* rstd = mean + NDIM;

  binarize_fp8_kernel<<<4096, 256, 0, stream>>>(x, (unsigned int*)xs, BDIM * KDIM / 4);
  binarize_fp8_kernel<<<4096, 256, 0, stream>>>(w, (unsigned int*)wss, NDIM * KDIM / 4);

  dim3 grid(NDIM / TN, BDIM / TM);
  bgemm_fp8_wmma_kernel<<<grid, 256, 0, stream>>>(xs, wss, y);

  stats_kernel<<<NDIM / 32, 256, 0, stream>>>(y, mean, rstd);

  norm_kernel<<<(BDIM * (size_t)NDIM / 4) / 256, 256, 0, stream>>>(y, mean, rstd, gamma, beta);
}
